// TransConvLayer_15126874817101
// MI455X (gfx1250) — matrix operand order repaired and checked by reading, not compiled, not run on hardware
//
#include <hip/hip_runtime.h>
#include <stdint.h>

#define DI __device__ __forceinline__

typedef __attribute__((ext_vector_type(16))) __bf16          v16bf;
typedef __attribute__((ext_vector_type(16))) unsigned short  v16us;
typedef __attribute__((ext_vector_type(8)))  unsigned short  v8us;
typedef __attribute__((ext_vector_type(4)))  unsigned short  v4us;
typedef __attribute__((ext_vector_type(8)))  float           v8f;

namespace {
constexpr int kN       = 65536;   // nodes
constexpr int kC       = 512;     // IN_CH
constexpr int kD       = 128;     // OUT_CH per head
constexpr int kH       = 8;       // HEADS
constexpr int kO       = kD * kH; // 1024
constexpr int kRows    = 64;      // node rows per workgroup
constexpr int kThreads = 128;     // 4 waves (wave32)
constexpr float kNf    = 65536.0f;

// workspace layout in 4-byte words; bf16 planes are planar hi / planar lo
constexpr size_t kWPl    = (size_t)kO * kC / 2;            // words per weight plane
constexpr size_t W_WQH   = 0;
constexpr size_t W_WQL   = W_WQH + kWPl;
constexpr size_t W_WKH   = W_WQL + kWPl;
constexpr size_t W_WKL   = W_WKH + kWPl;
constexpr size_t W_WVH   = W_WKL + kWPl;
constexpr size_t W_WVL   = W_WVH + kWPl;
constexpr size_t W_KVS   = W_WVL + kWPl;                   // fp32 accum [8][128][128]
constexpr size_t W_KSUM  = W_KVS + (size_t)kH * kD * kD;   // fp32 [8][128]
constexpr size_t W_SQQ   = W_KSUM + (size_t)kH * kD;
constexpr size_t W_SQK   = W_SQQ + 1;
constexpr size_t W_KVBH  = W_SQK + 7;                      // 32B aligned; bf16 hi plane of scaled kvs (B-frag order)
constexpr size_t W_KVBL  = W_KVBH + (size_t)kH * kD * kD / 2;
constexpr size_t W_KSUMS = W_KVBL + (size_t)kH * kD * kD / 2;  // scaled fp32 ksum
} // namespace

// ---------- bf16 split helpers ----------
DI unsigned bf16_rne(float x) {
  unsigned u = __float_as_uint(x);
  return (u + 0x7fffu + ((u >> 16) & 1u)) >> 16;
}
DI void split2(float x, unsigned short& h, unsigned short& l) {
  unsigned hu = bf16_rne(x);
  float hf = __uint_as_float(hu << 16);
  l = (unsigned short)bf16_rne(x - hf);
  h = (unsigned short)hu;
}
DI float b2f(unsigned short u) { return __uint_as_float(((unsigned)u) << 16); }
DI v8f zero8() {
  v8f z;
  #pragma unroll
  for (int i = 0; i < 8; ++i) z[i] = 0.0f;
  return z;
}

// ---------- fragment loads (contiguous k = 16*half + s, same order in A and B) ----------
// A (16x32 bf16): per lane, 16 contiguous K-values: K = half*16+0..15.
// p points at (row base + kBase) in a planar bf16 array.
DI v16bf ldA(const unsigned short* p, int half) {
  v8us a = *(const v8us*)(p + half * 16);
  v8us b = *(const v8us*)(p + half * 16 + 8);
  v16us r = __builtin_shufflevector(a, b, 0, 1, 2, 3, 4, 5, 6, 7,
                                    8, 9, 10, 11, 12, 13, 14, 15);
  return __builtin_bit_cast(v16bf, r);
}
// B (32x16 bf16): per lane, 16 contiguous K-values for its column.
DI v16bf ld16(const unsigned short* p) {
  return __builtin_bit_cast(v16bf, *(const v16us*)p);
}

// 3-term split FMA: C += Ahi*Bhi + Ahi*Blo + Alo*Bhi
DI v8f mma_split(v8f c, v16bf ah, v16bf al, v16bf bh, v16bf bl) {
  c = __builtin_amdgcn_wmma_f32_16x16x32_bf16(false, ah, false, bh, (short)0, c, false, false);
  c = __builtin_amdgcn_wmma_f32_16x16x32_bf16(false, ah, false, bl, (short)0, c, false, false);
  c = __builtin_amdgcn_wmma_f32_16x16x32_bf16(false, al, false, bh, (short)0, c, false, false);
  return c;
}

// cooperative load of a 64x512 fp32 tile -> planar hi/lo bf16 LDS planes
DI void load_x_tile(const float* X, int row0, unsigned short* Xh, unsigned short* Xl, int tid) {
  const float4* src = (const float4*)(X + (size_t)row0 * kC);
  for (int i = tid; i < kRows * kC / 4; i += kThreads) {
    float4 v = src[i];
    unsigned short h0, l0, h1, l1, h2, l2, h3, l3;
    split2(v.x, h0, l0);
    split2(v.y, h1, l1);
    split2(v.z, h2, l2);
    split2(v.w, h3, l3);
    v4us h, l;
    h[0] = h0; h[1] = h1; h[2] = h2; h[3] = h3;
    l[0] = l0; l[1] = l1; l[2] = l2; l[3] = l3;
    *(v4us*)(Xh + i * 4) = h;
    *(v4us*)(Xl + i * 4) = l;
  }
}

// one head: Y[16 rows x 128 cols] = X(16x512) * W^T + bias, into acc[8] C-tiles
DI void gemm_head(const unsigned short* Xh, const unsigned short* Xl,
                  int rowLocal, int half, int lane15,
                  const unsigned short* Wh, const unsigned short* Wl,
                  const float* bias, int h, v8f acc[8]) {
  #pragma unroll
  for (int t = 0; t < 8; ++t) acc[t] = zero8();
  #pragma unroll 1
  for (int kc = 0; kc < kC / 32; ++kc) {
    const unsigned short* pa = Xh + rowLocal * kC + kc * 32;
    const unsigned short* pal = Xl + rowLocal * kC + kc * 32;
    v16bf ah = ldA(pa, half), al = ldA(pal, half);
    #pragma unroll
    for (int t = 0; t < 8; ++t) {
      int off = (h * kD + t * 16 + lane15) * kC + kc * 32 + half * 16;
      v16bf bh = ld16(Wh + off);
      v16bf bl = ld16(Wl + off);
      acc[t] = mma_split(acc[t], ah, al, bh, bl);
    }
  }
  #pragma unroll
  for (int t = 0; t < 8; ++t) {
    float bb = bias[h * kD + t * 16 + lane15];
    #pragma unroll
    for (int v = 0; v < 8; ++v) acc[t][v] += bb;
  }
}

DI void block_reduce_add(float* scratch, int tid, float v, float* dst) {
  scratch[tid] = v;
  __syncthreads();
  for (int s = kThreads / 2; s > 0; s >>= 1) {
    if (tid < s) scratch[tid] += scratch[tid + s];
    __syncthreads();
  }
  if (tid == 0) atomicAdd(dst, scratch[0]);
}

// ---------------- kernels ----------------
__global__ void k_prep(const float* wq, const float* wk, const float* wv, unsigned* ws) {
  const int WN2 = kO * kC / 2;   // word pairs per weight
  int stride = gridDim.x * blockDim.x;
  int i0 = blockIdx.x * blockDim.x + threadIdx.x;
  unsigned short h0, l0, h1, l1;
  for (int j = i0; j < WN2; j += stride) {
    split2(wq[2 * j], h0, l0); split2(wq[2 * j + 1], h1, l1);
    ws[W_WQH + j] = (unsigned)h0 | ((unsigned)h1 << 16);
    ws[W_WQL + j] = (unsigned)l0 | ((unsigned)l1 << 16);
    split2(wk[2 * j], h0, l0); split2(wk[2 * j + 1], h1, l1);
    ws[W_WKH + j] = (unsigned)h0 | ((unsigned)h1 << 16);
    ws[W_WKL + j] = (unsigned)l0 | ((unsigned)l1 << 16);
    split2(wv[2 * j], h0, l0); split2(wv[2 * j + 1], h1, l1);
    ws[W_WVH + j] = (unsigned)h0 | ((unsigned)h1 << 16);
    ws[W_WVL + j] = (unsigned)l0 | ((unsigned)l1 << 16);
  }
  const int ZN = (int)(W_KVBH - W_KVS);
  float* wsf = (float*)ws;
  for (int j = i0; j < ZN; j += stride) wsf[W_KVS + j] = 0.0f;
}

__global__ void k_qnorm(const float* __restrict__ xq,
                        const unsigned short* __restrict__ wqH,
                        const unsigned short* __restrict__ wqL,
                        const float* __restrict__ bq, float* sumsq_q) {
  extern __shared__ unsigned short smem[];
  unsigned short* Xh = smem;              // 32768 us
  unsigned short* Xl = Xh + kRows * kC;   // 32768 us
  int tid = threadIdx.x, row0 = blockIdx.x * kRows;
  load_x_tile(xq, row0, Xh, Xl, tid);
  __syncthreads();
  int wave = tid >> 5, lane = tid & 31, half = lane >> 4, lane15 = lane & 15;
  int rowLocal = wave * 16 + lane15;
  float psum = 0.0f;
  #pragma unroll 1
  for (int h = 0; h < kH; ++h) {
    v8f acc[8];
    gemm_head(Xh, Xl, rowLocal, half, lane15, wqH, wqL, bq, h, acc);
    #pragma unroll
    for (int t = 0; t < 8; ++t) {
      #pragma unroll
      for (int v = 0; v < 8; ++v) psum += acc[t][v] * acc[t][v];
    }
  }
  __syncthreads();
  block_reduce_add((float*)smem, tid, psum, sumsq_q);
}

__global__ void k_kv(const float* __restrict__ xs,
                     const unsigned short* __restrict__ wkH, const unsigned short* __restrict__ wkL,
                     const float* __restrict__ bk,
                     const unsigned short* __restrict__ wvH, const unsigned short* __restrict__ wvL,
                     const float* __restrict__ bv,
                     float* kvs, float* ksum, float* sumsq_k) {
  extern __shared__ unsigned short smem[];
  unsigned short* Xh  = smem;                   // 32768 us
  unsigned short* Xl  = Xh + kRows * kC;        // 32768 us
  unsigned short* Kth = Xl + kRows * kC;        // K transposed [128 m][64 node], 8192 us
  unsigned short* Ktl = Kth + kD * kRows;       // 8192 us
  unsigned short* Vbh = Ktl + kD * kRows;       // V in B-frag order [(node>>4)][d][node&15], 8192 us
  unsigned short* Vbl = Vbh + kRows * kD;       // 8192 us
  int tid = threadIdx.x, row0 = blockIdx.x * kRows;
  load_x_tile(xs, row0, Xh, Xl, tid);
  __syncthreads();
  int wave = tid >> 5, lane = tid & 31, half = lane >> 4, lane15 = lane & 15;
  int rowLocal = wave * 16 + lane15;
  float psum = 0.0f;
  #pragma unroll 1
  for (int h = 0; h < kH; ++h) {
    v8f ka[8], va[8];
    gemm_head(Xh, Xl, rowLocal, half, lane15, wkH, wkL, bk, h, ka);
    gemm_head(Xh, Xl, rowLocal, half, lane15, wvH, wvL, bv, h, va);
    // stats + stage K (transposed) and V (B-frag order) to LDS
    #pragma unroll
    for (int t = 0; t < 8; ++t) {
      #pragma unroll
      for (int v = 0; v < 8; ++v) {
        float kx = ka[t][v];
        psum += kx * kx;
        int r = wave * 16 + v + 8 * half;   // node within tile
        int c = t * 16 + lane15;            // output column (m or d)
        unsigned short hh, ll;
        split2(kx, hh, ll);
        Kth[c * kRows + r] = hh; Ktl[c * kRows + r] = ll;
        split2(va[t][v], hh, ll);
        int dv = (wave * kD + c) * 16 + (v + 8 * half);
        Vbh[dv] = hh; Vbl[dv] = ll;
      }
    }
    __syncthreads();
    // column sums of K over the 64 staged nodes
    {
      int m = tid;  // 0..127
      float s = 0.0f;
      for (int n = 0; n < kRows; ++n)
        s += b2f(Kth[m * kRows + n]) + b2f(Ktl[m * kRows + n]);
      atomicAdd(&ksum[h * kD + m], s);
    }
    // K^T V via WMMA: contraction over 64 nodes, 64 tiles of 16x16, 16 per wave
    #pragma unroll 1
    for (int q = 0; q < 16; ++q) {
      int idx = wave * 16 + q;
      int tm = idx >> 3, td = idx & 7;
      int m    = tm * 16 + lane15;
      int dcol = td * 16 + lane15;
      v8f c = zero8();
      #pragma unroll
      for (int kc = 0; kc < 2; ++kc) {
        v16bf ah = ldA(Kth + m * kRows + kc * 32, half);
        v16bf al = ldA(Ktl + m * kRows + kc * 32, half);
        int ob = ((kc * 2 + half) * kD + dcol) * 16;
        v16bf bh = ld16(Vbh + ob);
        v16bf bl = ld16(Vbl + ob);
        c = mma_split(c, ah, al, bh, bl);
      }
      #pragma unroll
      for (int v = 0; v < 8; ++v) {
        int mm = tm * 16 + v + 8 * half;
        atomicAdd(&kvs[h * kD * kD + mm * kD + td * 16 + lane15], c[v]);
      }
    }
    __syncthreads();
  }
  block_reduce_add((float*)Kth, tid, psum, sumsq_k);
}

__global__ void k_finalize(const float* kvs, const float* ksum,
                           const float* sqq, const float* sqk,
                           unsigned short* kvbH, unsigned short* kvbL, float* ksumS) {
  float s = 1.0f / (sqrtf(*sqq) * sqrtf(*sqk));
  int stride = gridDim.x * blockDim.x;
  int i0 = blockIdx.x * blockDim.x + threadIdx.x;
  // rewrite scaled kvs into B-fragment order: [(h*4+kc)*2+half][d][k&15]
  for (int j = i0; j < kH * kD * kD; j += stride) {
    int h = j >> 14, m = (j >> 7) & 127, d = j & 127;
    int kc = m >> 5, hf = (m >> 4) & 1, kk = m & 15;
    int dst = (((h * 4 + kc) * 2 + hf) * kD + d) * 16 + kk;
    unsigned short hh, ll;
    split2(kvs[j] * s, hh, ll);
    kvbH[dst] = hh; kvbL[dst] = ll;
  }
  for (int j = i0; j < kH * kD; j += stride) ksumS[j] = ksum[j] * s;
}

__global__ void k_out(const float* __restrict__ xq, const float* __restrict__ xs,
                      const unsigned short* __restrict__ wqH, const unsigned short* __restrict__ wqL,
                      const float* __restrict__ bq,
                      const unsigned short* __restrict__ wvH, const unsigned short* __restrict__ wvL,
                      const float* __restrict__ bv,
                      const unsigned short* __restrict__ kvbH, const unsigned short* __restrict__ kvbL,
                      const float* __restrict__ ksumS,
                      float* __restrict__ out) {
  extern __shared__ unsigned short smem[];
  unsigned short* Xqh = smem;                 // 32768 us
  unsigned short* Xql = Xqh + kRows * kC;
  unsigned short* Xsh = Xql + kRows * kC;
  unsigned short* Xsl = Xsh + kRows * kC;
  unsigned short* Qh  = Xsl + kRows * kC;     // 4 waves x [16][128], 8192 us
  unsigned short* Ql  = Qh + 4 * 16 * kD;     // 8192 us
  float* nrm = (float*)(Ql + 4 * 16 * kD);    // 64 floats
  int tid = threadIdx.x, row0 = blockIdx.x * kRows;
  load_x_tile(xq, row0, Xqh, Xql, tid);
  load_x_tile(xs, row0, Xsh, Xsl, tid);
  __syncthreads();
  int wave = tid >> 5, lane = tid & 31, half = lane >> 4, lane15 = lane & 15;
  int rowLocal = wave * 16 + lane15;
  unsigned short* Qwh = Qh + wave * 16 * kD;  // wave-private Q staging [16][128]
  unsigned short* Qwl = Ql + wave * 16 * kD;
  v8f outacc[8];
  #pragma unroll
  for (int t = 0; t < 8; ++t) outacc[t] = zero8();

  #pragma unroll 1
  for (int h = 0; h < kH; ++h) {
    // Q tile for this head -> LDS planes (registers freed afterwards)
    {
      v8f qa[8];
      gemm_head(Xqh, Xql, rowLocal, half, lane15, wqH, wqL, bq, h, qa);
      #pragma unroll
      for (int t = 0; t < 8; ++t) {
        #pragma unroll
        for (int v = 0; v < 8; ++v) {
          unsigned short hh, ll;
          split2(qa[t][v], hh, ll);
          int dq = (v + 8 * half) * kD + t * 16 + lane15;
          Qwh[dq] = hh; Qwl[dq] = ll;
        }
      }
    }
    // V tile for this head (kept in registers)
    v8f va[8];
    gemm_head(Xsh, Xsl, rowLocal, half, lane15, wvH, wvL, bv, h, va);
    // normalizer: Q . ksum_scaled (lanes 0-15, one row each; wave-private)
    if (lane < 16) {
      const float* kp = ksumS + h * kD;
      float s = 0.0f;
      for (int m = 0; m < kD; ++m)
        s += (b2f(Qwh[lane * kD + m]) + b2f(Qwl[lane * kD + m])) * kp[m];
      nrm[wave * 16 + lane] = s;
    }
    // num = Q @ kvs_scaled, then fold (num + N*V)/(nrm + N) into outacc
    #pragma unroll 1
    for (int t = 0; t < 8; ++t) {
      v8f c = zero8();
      #pragma unroll 1
      for (int kc = 0; kc < 4; ++kc) {
        v16bf ah = ldA(Qwh + lane15 * kD + kc * 32, half);
        v16bf al = ldA(Qwl + lane15 * kD + kc * 32, half);
        int ob = (((h * 4 + kc) * 2 + half) * kD + t * 16 + lane15) * 16;
        v16bf bh = ld16(kvbH + ob);
        v16bf bl = ld16(kvbL + ob);
        c = mma_split(c, ah, al, bh, bl);
      }
      #pragma unroll
      for (int v = 0; v < 8; ++v) {
        float nv = nrm[wave * 16 + v + 8 * half];
        outacc[t][v] += (c[v] + kNf * va[t][v]) / (nv + kNf);
      }
    }
  }
  // head mean, write out [n][128]
  #pragma unroll
  for (int t = 0; t < 8; ++t) {
    #pragma unroll
    for (int v = 0; v < 8; ++v) {
      int n = row0 + wave * 16 + v + 8 * half;
      out[(size_t)n * kD + t * 16 + lane15] = outacc[t][v] * 0.125f;
    }
  }
}

extern "C" void kernel_launch(void* const* d_in, const int* in_sizes, int n_in,
                              void* d_out, int out_size, void* d_ws, size_t ws_size,
                              hipStream_t stream) {
  const float* xq = (const float*)d_in[0];
  const float* xs = (const float*)d_in[1];
  const float* wq = (const float*)d_in[2];
  const float* bq = (const float*)d_in[3];
  const float* wk = (const float*)d_in[4];
  const float* bk = (const float*)d_in[5];
  const float* wv = (const float*)d_in[6];
  const float* bv = (const float*)d_in[7];
  unsigned* ws = (unsigned*)d_ws;
  float* wsf = (float*)d_ws;
  float* out = (float*)d_out;
  const unsigned short* wqH = (const unsigned short*)(ws + W_WQH);
  const unsigned short* wqL = (const unsigned short*)(ws + W_WQL);
  const unsigned short* wkH = (const unsigned short*)(ws + W_WKH);
  const unsigned short* wkL = (const unsigned short*)(ws + W_WKL);
  const unsigned short* wvH = (const unsigned short*)(ws + W_WVH);
  const unsigned short* wvL = (const unsigned short*)(ws + W_WVL);
  unsigned short* kvbH = (unsigned short*)(ws + W_KVBH);
  unsigned short* kvbL = (unsigned short*)(ws + W_KVBL);

  const int lds_qnorm = 2 * kRows * kC * 2;                       // 128 KB
  const int lds_kv    = (2 * kRows * kC + 4 * kRows * kD) * 2;    // 192 KB
  const int lds_out   = (4 * kRows * kC + 2 * 4 * 16 * kD) * 2 + 64 * 4;  // ~288 KB
  (void)hipFuncSetAttribute((const void*)k_qnorm, hipFuncAttributeMaxDynamicSharedMemorySize, lds_qnorm);
  (void)hipFuncSetAttribute((const void*)k_kv,    hipFuncAttributeMaxDynamicSharedMemorySize, lds_kv);
  (void)hipFuncSetAttribute((const void*)k_out,   hipFuncAttributeMaxDynamicSharedMemorySize, lds_out);

  k_prep<<<2048, 256, 0, stream>>>(wq, wk, wv, ws);
  k_qnorm<<<kN / kRows, kThreads, lds_qnorm, stream>>>(xq, wqH, wqL, bq, wsf + W_SQQ);
  k_kv<<<kN / kRows, kThreads, lds_kv, stream>>>(xs, wkH, wkL, bk, wvH, wvL, bv,
                                                 wsf + W_KVS, wsf + W_KSUM, wsf + W_SQK);
  k_finalize<<<512, 256, 0, stream>>>(wsf + W_KVS, wsf + W_KSUM, wsf + W_SQQ, wsf + W_SQK,
                                      kvbH, kvbL, wsf + W_KSUMS);
  k_out<<<kN / kRows, kThreads, lds_out, stream>>>(xq, xs, wqH, wqL, bq, wvH, wvL, bv,
                                                   kvbH, kvbL, wsf + W_KSUMS, out);
}
